// NNShotModel_52261162058397
// MI455X (gfx1250) — compile-verified, matrix operand order checked by reading, not run on hardware
//
#include <hip/hip_runtime.h>
#include <hip/hip_bf16.h>

typedef __attribute__((ext_vector_type(16))) __bf16 v16bf;
typedef __attribute__((ext_vector_type(8)))  float  v8f;
typedef __attribute__((ext_vector_type(4)))  __bf16 bf16x4;

#define H 128
#define NLAB 64
#define NEG_BIG (-1.0e9f)
#define RPW 32        // query rows per wave (two 16-row A tiles)
#define RPB 256       // rows per block (8 waves)
#define CHTILES 8     // 16-col tiles per staged chunk
#define CHCOLS 128    // columns per staged chunk (32 KB bf16)
#define COLB 16       // column-blocks (grid.y)

union BFrag { v16bf v; uint4 u[2]; };

// order-preserving f32 -> u32 key (monotone for all floats)
__device__ __forceinline__ unsigned encf(float x) {
    unsigned b = __float_as_uint(x);
    return (b & 0x80000000u) ? ~b : (b | 0x80000000u);
}
__device__ __forceinline__ float decf(unsigned u) {
    unsigned b = (u & 0x80000000u) ? (u & 0x7fffffffu) : ~u;
    return __uint_as_float(b);
}

// cheap max: v_cmp + v_cndmask, avoids llvm.maxnum's canonicalize self-max ops
__device__ __forceinline__ float fmaxq(float a, float b) { return a > b ? a : b; }

// ---------------- preprocessing kernels ----------------

__global__ void k_zero(int* counts) {
    if (threadIdx.x < NLAB) counts[threadIdx.x] = 0;
}

__global__ void k_hist(const int* __restrict__ lsup, int* __restrict__ counts, int M) {
    int i = blockIdx.x * blockDim.x + threadIdx.x;
    if (i < M) atomicAdd(&counts[lsup[i]], 1);
}

// serial scan over 64 bins; segments padded to multiples of 16 cols. segBase has 65 entries.
__global__ void k_scan(const int* __restrict__ counts, int* __restrict__ segBase,
                       int* __restrict__ cursor) {
    if (threadIdx.x == 0 && blockIdx.x == 0) {
        int base = 0;
        for (int l = 0; l < NLAB; ++l) {
            segBase[l] = base;
            cursor[l]  = base;
            base += ((counts[l] + 15) >> 4) << 4;
        }
        segBase[NLAB] = base;   // P = total padded columns
    }
}

// zero padded tail rows so WMMA reads clean data
__global__ void k_padfill(const int* __restrict__ counts, const int* __restrict__ segBase,
                          __bf16* __restrict__ ss) {
    int l = blockIdx.x;
    int start = segBase[l] + counts[l];
    int end   = segBase[l + 1];
    for (int s = start + (int)threadIdx.x; s < end; s += (int)blockDim.x) {
        __bf16* row = ss + (size_t)s * H;
        for (int i = 0; i < H; ++i) row[i] = (__bf16)0.0f;
    }
}

// L2-normalize support rows, convert to bf16, scatter into label-sorted order
__global__ __launch_bounds__(256) void k_scatter(const float* __restrict__ sup,
                                                 const int* __restrict__ lsup,
                                                 __bf16* __restrict__ ss,
                                                 int* __restrict__ cursor, int M) {
    int row  = blockIdx.x * 8 + (threadIdx.x >> 5);
    int lane = threadIdx.x & 31;
    if (row >= M) return;
    const float* src = sup + (size_t)row * H;
    float4 x = *(const float4*)(src + lane * 4);
    float s2 = x.x*x.x + x.y*x.y + x.z*x.z + x.w*x.w;
    #pragma unroll
    for (int off = 1; off < 32; off <<= 1) s2 += __shfl_xor(s2, off, 32);
    float scale = 1.0f / fmaxf(sqrtf(s2), 1e-12f);
    int lab = lsup[row];
    int pos = 0;
    if (lane == 0) pos = atomicAdd(&cursor[lab], 1);
    pos = __shfl(pos, 0, 32);
    bf16x4 p;
    p[0] = (__bf16)(x.x * scale); p[1] = (__bf16)(x.y * scale);
    p[2] = (__bf16)(x.z * scale); p[3] = (__bf16)(x.w * scale);
    *(bf16x4*)(ss + (size_t)pos * H + lane * 4) = p;
}

// L2-normalize query rows, convert to bf16 (original order)
__global__ __launch_bounds__(256) void k_qnorm(const float* __restrict__ q,
                                               __bf16* __restrict__ qn, int N) {
    int row  = blockIdx.x * 8 + (threadIdx.x >> 5);
    int lane = threadIdx.x & 31;
    if (row >= N) return;
    const float* src = q + (size_t)row * H;
    float4 x = *(const float4*)(src + lane * 4);
    float s2 = x.x*x.x + x.y*x.y + x.z*x.z + x.w*x.w;
    #pragma unroll
    for (int off = 1; off < 32; off <<= 1) s2 += __shfl_xor(s2, off, 32);
    float scale = 1.0f / fmaxf(sqrtf(s2), 1e-12f);
    bf16x4 p;
    p[0] = (__bf16)(x.x * scale); p[1] = (__bf16)(x.y * scale);
    p[2] = (__bf16)(x.z * scale); p[3] = (__bf16)(x.w * scale);
    *(bf16x4*)(qn + (size_t)row * H + lane * 4) = p;
}

__global__ void k_initnear(unsigned* nw, int n) {
    int i = blockIdx.x * blockDim.x + threadIdx.x;
    if (i < n) nw[i] = encf(NEG_BIG);
}

// ---------------- fused WMMA GEMM + segment-max ----------------
// grid = (N/256 row-blocks, COLB col-blocks). Block = 8 waves; each wave owns 32
// query rows (two A tiles resident in VGPRs) and sweeps this block's column
// chunks. B staged once per block into LDS with async global->LDS copies
// (32 KB chunks, double-buffered), shared by all 8 waves. Partial per-(row,label)
// maxima merged across col-blocks with atomicMax on order-preserving u32 keys.
__global__ __launch_bounds__(256) void k_main(const __bf16* __restrict__ qn,
                                              const __bf16* __restrict__ ss,
                                              const int* __restrict__ segBase,
                                              const int* __restrict__ counts,
                                              unsigned* __restrict__ nearWork,
                                              int capChunks, int cpb) {
    __shared__ __align__(32) __bf16 sB[2][CHCOLS][H];   // 64 KB double buffer

    const int lane    = threadIdx.x & 31;
    const int wave    = threadIdx.x >> 5;
    const int rowBase = blockIdx.x * RPB + wave * RPW;
    const int hi      = lane >> 4;
    const int l15     = lane & 15;

    const int cA = blockIdx.y * cpb;
    const int cB = min(cA + cpb, capChunks);

    const int P = segBase[NLAB];
    const int T = P >> 4;                     // total real 16-col tiles

    // A fragments: 32 rows x K=128 (ISA 16-bit A layout: lane<16 K{0..7,16..23},
    // lane>=16 K{8..15,24..31} per 32-wide k-step)
    BFrag a0[4], a1[4];
    {
        const __bf16* q0 = qn + (size_t)(rowBase + l15) * H;
        const __bf16* q1 = qn + (size_t)(rowBase + 16 + l15) * H;
        #pragma unroll
        for (int k = 0; k < 4; ++k) {
            int k0 = k * 32 + hi * 8;
            a0[k].u[0] = *(const uint4*)(q0 + k0);
            a0[k].u[1] = *(const uint4*)(q0 + k0 + 16);
            a1[k].u[0] = *(const uint4*)(q1 + k0);
            a1[k].u[1] = *(const uint4*)(q1 + k0 + 16);
        }
    }

    // locate segment containing this block's first column
    int seg = 0;
    const int startCol = cA * CHCOLS;
    while (seg < NLAB && segBase[seg + 1] <= startCol) seg++;
    int segEnd = (seg < NLAB) ? segBase[seg] + counts[seg] : 0x7fffffff;

    float m[16];
    #pragma unroll
    for (int r = 0; r < 16; ++r) m[r] = -1.0e30f;

    auto stageChunk = [&](int c, int b) {
        const __bf16* src0 = ss + (size_t)c * CHCOLS * H;
        #pragma unroll
        for (int r = 0; r < 8; ++r) {
            int idx = r * 256 + (int)threadIdx.x;      // 2048 x 16B = 32 KB
            int col = idx >> 4;
            int ko  = (idx & 15) * 8;
            const __bf16* g = src0 + (size_t)col * H + ko;
            unsigned loff = (unsigned)(uintptr_t)(&sB[b][col][ko]);
            asm volatile("global_load_async_to_lds_b128 %0, %1, off"
                         :: "v"(loff), "v"(g) : "memory");
        }
    };

    auto finalize = [&](int sg) {
        #pragma unroll
        for (int r = 0; r < 16; ++r) {
            #pragma unroll
            for (int off = 1; off < 16; off <<= 1)
                m[r] = fmaxq(m[r], __shfl_xor(m[r], off, 32));
        }
        if (l15 == 0 && sg != 0) {                      // label 0 == PAD: never emit
            #pragma unroll
            for (int rt = 0; rt < 2; ++rt)
                #pragma unroll
                for (int g = 0; g < 8; ++g) {
                    float mv = m[rt * 8 + g];
                    if (mv > -1.0e29f) {
                        int row = rowBase + rt * 16 + hi * 8 + g;
                        atomicMax(&nearWork[(size_t)row * NLAB + sg],
                                  encf(2.0f * mv - 2.0f));
                    }
                }
        }
        #pragma unroll
        for (int r = 0; r < 16; ++r) m[r] = -1.0e30f;
    };

    auto computeTile = [&](int i, int b, int jt) {
        BFrag bb[4];
        #pragma unroll
        for (int k = 0; k < 4; ++k) {
            int k0 = k * 32 + hi * 8;
            const __bf16* p = &sB[b][jt * 16 + l15][0];
            bb[k].u[0] = *(const uint4*)(p + k0);
            bb[k].u[1] = *(const uint4*)(p + k0 + 16);
        }
        v8f c0 = {}, c1 = {};
        #pragma unroll
        for (int k = 0; k < 4; ++k)
            c0 = __builtin_amdgcn_wmma_f32_16x16x32_bf16(false, a0[k].v, false, bb[k].v,
                                                         (short)0, c0, false, false);
        #pragma unroll
        for (int k = 0; k < 4; ++k)
            c1 = __builtin_amdgcn_wmma_f32_16x16x32_bf16(false, a1[k].v, false, bb[k].v,
                                                         (short)0, c1, false, false);
        if ((i + 1) * 16 <= segEnd) {
            // interior tile: no padding possible -> pure cmp+select max
            #pragma unroll
            for (int g = 0; g < 8; ++g) {
                m[g]     = fmaxq(m[g],     c0[g]);
                m[8 + g] = fmaxq(m[8 + g], c1[g]);
            }
        } else {
            // tail tile of a segment: mask padded columns
            const bool pad = (i * 16 + l15) >= segEnd;
            #pragma unroll
            for (int g = 0; g < 8; ++g) {
                float v0 = pad ? -1.0e30f : c0[g];
                float v1 = pad ? -1.0e30f : c1[g];
                m[g]     = fmaxq(m[g],     v0);
                m[8 + g] = fmaxq(m[8 + g], v1);
            }
        }
    };

    int buf = 0;
    if (cA < cB) {
        stageChunk(cA, 0);
        asm volatile("s_wait_asynccnt 0" ::: "memory");
        __syncthreads();
    }
    for (int c = cA; c < cB; ++c) {
        if (c + 1 < cB) stageChunk(c + 1, buf ^ 1);     // prefetch next chunk
        #pragma unroll
        for (int j = 0; j < CHTILES; ++j) {
            int i = c * CHTILES + j;
            if (i < T) computeTile(i, buf, j);
            int done = (i + 1) * 16;
            while (seg < NLAB && done >= segBase[seg + 1]) {
                finalize(seg);
                seg++;
                segEnd = (seg < NLAB) ? segBase[seg] + counts[seg] : 0x7fffffff;
            }
        }
        asm volatile("s_wait_asynccnt 0" ::: "memory");
        __syncthreads();
        buf ^= 1;
    }
    if (seg < NLAB) finalize(seg);                      // partial segment at range end
}

// ---------------- final pass: decode, mask, argmax-label ----------------
__global__ void k_best(const unsigned* __restrict__ nearWork,
                       const int* __restrict__ qids, const int* __restrict__ lsup,
                       float* __restrict__ outBest, float* __restrict__ outNear, int N) {
    int row = blockIdx.x * blockDim.x + threadIdx.x;
    if (row >= N) return;
    const bool qv = (qids[row] != 0);
    const int label0 = lsup[0];
    float bv = -3.0e38f; int bl = label0;
    for (int l = 0; l < NLAB; ++l) {
        float v = qv ? decf(nearWork[(size_t)row * NLAB + l]) : NEG_BIG;
        outNear[(size_t)row * NLAB + l] = v;
        if (v > bv) { bv = v; bl = l; }
    }
    outBest[row] = qv ? (float)bl : (float)label0;
}

// ---------------- launcher ----------------

extern "C" void kernel_launch(void* const* d_in, const int* in_sizes, int n_in,
                              void* d_out, int out_size, void* d_ws, size_t ws_size,
                              hipStream_t stream) {
    const float* q    = (const float*)d_in[0];
    const float* sup  = (const float*)d_in[1];
    const int*   qids = (const int*)d_in[2];
    const int*   lsup = (const int*)d_in[3];
    const int N = in_sizes[0] / H;     // 8192
    const int M = in_sizes[1] / H;     // 16384

    const int capCols   = ((M + NLAB * 16 + CHCOLS - 1) / CHCOLS) * CHCOLS;
    const int capChunks = capCols / CHCOLS;
    const int cpb       = (capChunks + COLB - 1) / COLB;

    auto align256 = [](size_t x) { return (x + 255) & ~(size_t)255; };
    char* ws = (char*)d_ws;
    size_t off = 0;
    __bf16* qn      = (__bf16*)(ws + off); off = align256(off + (size_t)N * H * 2);
    __bf16* ssorted = (__bf16*)(ws + off); off = align256(off + (size_t)capCols * H * 2);
    int* counts  = (int*)(ws + off); off = align256(off + NLAB * 4);
    int* segBase = (int*)(ws + off); off = align256(off + (NLAB + 1) * 4);
    int* cursor  = (int*)(ws + off); off = align256(off + NLAB * 4);
    unsigned* nearWork = (unsigned*)(ws + off); off = align256(off + (size_t)N * NLAB * 4);
    (void)ws_size; (void)n_in; (void)out_size;

    float* outBest = (float*)d_out;      // [N] best labels (as float)
    float* outNear = outBest + N;        // [N, 64] nearest

    k_zero<<<1, 64, 0, stream>>>(counts);
    k_hist<<<(M + 255) / 256, 256, 0, stream>>>(lsup, counts, M);
    k_scan<<<1, 1, 0, stream>>>(counts, segBase, cursor);
    k_padfill<<<NLAB, 64, 0, stream>>>(counts, segBase, ssorted);
    k_scatter<<<(M + 7) / 8, 256, 0, stream>>>(sup, lsup, ssorted, cursor, M);
    k_qnorm<<<(N + 7) / 8, 256, 0, stream>>>(q, qn, N);
    k_initnear<<<(N * NLAB + 255) / 256, 256, 0, stream>>>(nearWork, N * NLAB);

    dim3 grid(N / RPB, COLB);
    k_main<<<grid, 256, 0, stream>>>(qn, ssorted, segBase, counts, nearWork,
                                     capChunks, cpb);

    k_best<<<(N + 255) / 256, 256, 0, stream>>>(nearWork, qids, lsup,
                                                outBest, outNear, N);
}
